// SpecformerZINC_59124519797409
// MI455X (gfx1250) — compile-verified, matrix operand order checked by reading, not run on hardware
//
#include <hip/hip_runtime.h>
#include <math.h>

#define BB 8
#define NN 128
#define HH 128
#define NHEAD 4
#define DHD 32
#define LL 4
#define EPSF 1e-5f

typedef __attribute__((ext_vector_type(16))) _Float16 v16h;
typedef __attribute__((ext_vector_type(8)))  float    v8f;
typedef _Float16 half_t;

__device__ inline float geluf(float x) {
  return 0.5f * x * (1.0f + erff(x * 0.70710678118654752f));
}

// Load a 16x32 f16 tile (row-major in LDS, leading dim ld) into a WMMA
// A/B fragment per the CDNA5 16-bit 16x32 layout:
// lane l: row = l&15 ; VGPR i holds K pair k = (i>>2)*16 + (l>>4)*8 + (i&3)*2
__device__ inline v16h frag_ld(const half_t* sm, int ld, int lane) {
  v16h f;
  int r = lane & 15;
  int kh = (lane >> 4) << 3;
#pragma unroll
  for (int i = 0; i < 8; ++i) {
    int k = ((i >> 2) << 4) + kh + ((i & 3) << 1);
    f[2 * i]     = sm[r * ld + k];
    f[2 * i + 1] = sm[r * ld + k + 1];
  }
  return f;
}

// ---------------- node-level kernels (tiny, VALU) ----------------

// SineEncoding + eig = eeig @ eigw_W^T + b      grid: B*N, block: 128
__global__ void k_sine_eig(const float* __restrict__ e, const float* __restrict__ W,
                           const float* __restrict__ bias, float* __restrict__ eig) {
  int row = blockIdx.x;
  int h = threadIdx.x;
  __shared__ float ee[129];
  float ev = e[row];
  if (h < 64) {
    float dv = __expf((float)(2 * h) * (-9.210340371976184f / 128.0f));
    float pe = ev * 100.0f * dv;
    ee[1 + h]  = __sinf(pe);
    ee[65 + h] = __cosf(pe);
  }
  if (h == 0) ee[0] = ev;
  __syncthreads();
  float acc = bias[h];
  const float* wr = W + h * 129;
  for (int j = 0; j < 129; ++j) acc += ee[j] * wr[j];
  eig[row * HH + h] = acc;
}

// LayerNorm over last dim (128)                 grid: B*N, block: 128
__global__ void k_ln(const float* __restrict__ x, const float* __restrict__ g,
                     const float* __restrict__ b, float* __restrict__ y) {
  int row = blockIdx.x;
  int h = threadIdx.x;
  __shared__ float red[HH];
  float v = x[row * HH + h];
  red[h] = v;
  __syncthreads();
  for (int s = 64; s > 0; s >>= 1) { if (h < s) red[h] += red[h + s]; __syncthreads(); }
  float mu = red[0] / 128.0f;
  __syncthreads();
  float d = v - mu;
  red[h] = d * d;
  __syncthreads();
  for (int s = 64; s > 0; s >>= 1) { if (h < s) red[h] += red[h + s]; __syncthreads(); }
  float var = red[0] / 128.0f;
  y[row * HH + h] = d * rsqrtf(var + EPSF) * g[h] + b[h];
}

// out[row,n] = (opt act)(in_row . W[n,:] + bias[n]) + res   (K = 128)
// if mu!=null: input element k is first mapped relu((x-mu)*ivar*g+b) (fused BN)
__global__ void k_rowgemm(const float* __restrict__ in, const float* __restrict__ W,
                          const float* __restrict__ bias, const float* __restrict__ res,
                          float* __restrict__ out, int Nout, int act,
                          const float* __restrict__ mu, const float* __restrict__ ivar,
                          const float* __restrict__ bg, const float* __restrict__ bb) {
  int row = blockIdx.x;
  int t = threadIdx.x;
  __shared__ float rsh[HH];
  float v = in[row * HH + t];
  if (mu) v = fmaxf((v - mu[t]) * ivar[t] * bg[t] + bb[t], 0.0f);
  rsh[t] = v;
  __syncthreads();
  for (int n = t; n < Nout; n += HH) {
    float acc = bias[n];
    const float* wr = W + n * HH;
    for (int k = 0; k < HH; ++k) acc += rsh[k] * wr[k];
    if (act == 1) acc = geluf(acc);
    if (res) acc += res[row * Nout + n];
    out[row * Nout + n] = acc;
  }
}

// masked MHA over eigen tokens                grid: (NH, B), block: 128 (=query)
__global__ void k_attn(const float* __restrict__ qkv, const int* __restrict__ length,
                       float* __restrict__ attn_out, float* __restrict__ ctx) {
  int hd = blockIdx.x, b = blockIdx.y;
  int t = threadIdx.x;
  __shared__ float Ksh[NN][DHD + 1];
  __shared__ float Vsh[NN][DHD + 1];
  int len = length[b];
  const float* base = qkv + (size_t)(b * NN) * (3 * HH);
  for (int d = 0; d < DHD; ++d) {
    Ksh[t][d] = base[t * 3 * HH + HH     + hd * DHD + d];
    Vsh[t][d] = base[t * 3 * HH + 2 * HH + hd * DHD + d];
  }
  float q[DHD];
  for (int d = 0; d < DHD; ++d) q[d] = base[t * 3 * HH + hd * DHD + d];
  __syncthreads();
  const float scale = 0.17677669529663689f; // 1/sqrt(32)
  float mmax = -1e30f;
  for (int j = 0; j < len; ++j) {
    float s = 0.f;
    for (int d = 0; d < DHD; ++d) s += q[d] * Ksh[j][d];
    mmax = fmaxf(mmax, s * scale);
  }
  float den = 0.f, cacc[DHD];
  for (int d = 0; d < DHD; ++d) cacc[d] = 0.f;
  for (int j = 0; j < len; ++j) {
    float s = 0.f;
    for (int d = 0; d < DHD; ++d) s += q[d] * Ksh[j][d];
    float p = __expf(s * scale - mmax);
    den += p;
    for (int d = 0; d < DHD; ++d) cacc[d] += p * Vsh[j][d];
  }
  float inv = 1.0f / den;
  float* ao = attn_out + ((size_t)(b * NHEAD + hd) * NN + t) * NN;
  for (int j = 0; j < NN; ++j) {
    float p = 0.f;
    if (j < len) {
      float s = 0.f;
      for (int d = 0; d < DHD; ++d) s += q[d] * Ksh[j][d];
      p = __expf(s * scale - mmax) * inv;
    }
    ao[j] = p;
  }
  float* cp = ctx + (size_t)(b * NN + t) * HH + hd * DHD;
  for (int d = 0; d < DHD; ++d) cp[d] = cacc[d] * inv;
}

// new_e[b,k,n] = eig[b,n,:] . dec_W[k,:] + dec_b[k]   grid: B, block: 128
__global__ void k_dec(const float* __restrict__ eig, const float* __restrict__ W,
                      const float* __restrict__ bias, float* __restrict__ newe) {
  int b = blockIdx.x;
  int n = threadIdx.x;
  __shared__ float Wsh[NHEAD * HH];
  for (int i = n; i < NHEAD * HH; i += NN) Wsh[i] = W[i];
  __syncthreads();
  const float* er = eig + (size_t)(b * NN + n) * HH;
  for (int k = 0; k < NHEAD; ++k) {
    float acc = bias[k];
    for (int j = 0; j < HH; ++j) acc += er[j] * Wsh[k * HH + j];
    newe[(b * NHEAD + k) * NN + n] = acc;
  }
}

// ---------------- edge-level kernels (WMMA) ----------------

// filters[b,n,p,k] = sum_m U[n,m]*new_e[k,m]*U[p,m]   grid: (8,8,B*NH) block: 32
__global__ void k_filters(const float* __restrict__ u, const float* __restrict__ newe,
                          float* __restrict__ filt) {
  int pt = blockIdx.x, nt = blockIdx.y, bk = blockIdx.z;
  int b = bk >> 2, k = bk & 3;
  int lane = threadIdx.x;
  __shared__ half_t As[16][32];
  __shared__ half_t Bs[16][32];
  __shared__ float nev[32];
  v8f acc = {};
  int n0 = nt * 16, p0 = pt * 16;
  for (int mc = 0; mc < 4; ++mc) {
    int m0 = mc * 32;
    nev[lane] = newe[(b * NHEAD + k) * NN + m0 + lane];
    __syncthreads();
    int i = lane >> 1, c0 = (lane & 1) * 16;
    for (int c = 0; c < 16; ++c) {
      int mm = c0 + c;
      As[i][mm] = (half_t)(u[((size_t)(b * NN) + (n0 + i)) * NN + m0 + mm] * nev[mm]);
      Bs[i][mm] = (half_t)(u[((size_t)(b * NN) + (p0 + i)) * NN + m0 + mm]);
    }
    __syncthreads();
    v16h a  = frag_ld(&As[0][0], 32, lane);
    v16h bf = frag_ld(&Bs[0][0], 32, lane);
    acc = __builtin_amdgcn_wmma_f32_16x16x32_f16(false, a, false, bf, (short)0, acc,
                                                 false, false);
    __syncthreads();
  }
  int col = lane & 15, rb = (lane >> 4) * 8;
#pragma unroll
  for (int r = 0; r < 8; ++r) {
    int nrow = n0 + rb + r, pcol = p0 + col;
    filt[(((size_t)(b * NN) + nrow) * NN + pcol) * NHEAD + k] = acc[r];
  }
}

// bases = gelu(gelu([eye,filters]@fe1^T+b1)@fe2^T+b2), f16  grid: B*N*N/16 block: 256
__global__ void k_bases(const float* __restrict__ filt, const float* __restrict__ fe1W,
                        const float* __restrict__ fe1b, const float* __restrict__ fe2W,
                        const float* __restrict__ fe2b, half_t* __restrict__ bases) {
  int t = threadIdx.x;
  int r0 = blockIdx.x * 16; // global edge row base, edge = (b*128+i)*128+j
  __shared__ half_t Wsh[HH * HH];
  __shared__ half_t As[16 * HH];
  __shared__ float f1sh[HH * 5];
  __shared__ float f1bsh[HH];
  __shared__ float f2bsh[HH];
  __shared__ float frow[16][5];
  for (int i = t; i < HH * HH; i += 256) Wsh[i] = (half_t)fe2W[i];
  for (int i = t; i < HH * 5; i += 256) f1sh[i] = fe1W[i];
  if (t < HH) { f1bsh[t] = fe1b[t]; f2bsh[t] = fe2b[t]; }
  if (t < 64) {
    int row = t >> 2, k = t & 3;
    frow[row][1 + k] = filt[(size_t)(r0 + row) * NHEAD + k];
  }
  if (t < 16) {
    int edge = r0 + t;
    int i = (edge >> 7) & 127, j = edge & 127;
    frow[t][0] = (i == j) ? 1.0f : 0.0f;
  }
  __syncthreads();
  for (int idx = t; idx < 16 * HH; idx += 256) {
    int row = idx >> 7, h = idx & 127;
    const float* w = &f1sh[h * 5];
    float a = f1bsh[h] + frow[row][0] * w[0] + frow[row][1] * w[1] +
              frow[row][2] * w[2] + frow[row][3] * w[3] + frow[row][4] * w[4];
    As[row * HH + h] = (half_t)geluf(a);
  }
  __syncthreads();
  int wv = t >> 5, lane = t & 31;
  int n0 = wv * 16;
  v8f acc = {};
#pragma unroll
  for (int kc = 0; kc < 4; ++kc) {
    v16h a  = frag_ld(&As[kc * 32], HH, lane);
    v16h bf = frag_ld(&Wsh[n0 * HH + kc * 32], HH, lane);
    acc = __builtin_amdgcn_wmma_f32_16x16x32_f16(false, a, false, bf, (short)0, acc,
                                                 false, false);
  }
  int col = lane & 15, rb = (lane >> 4) * 8;
#pragma unroll
  for (int r = 0; r < 8; ++r) {
    int h = n0 + col;
    float v = geluf(acc[r] + f2bsh[h]);
    bases[(size_t)(r0 + rb + r) * HH + h] = (half_t)v;
  }
}

// per layer: aggr[b,j,:] = sum_{i<len} gelu((xf[b,i,:]+bond[ef])@pre^T+pb)*bases*mask
// grid: (8 jtiles, B), block: 256 (8 waves, wave w -> output cols w*16..)
__global__ void k_msg(const float* __restrict__ xf, const half_t* __restrict__ bases,
                      const float* __restrict__ preW, const float* __restrict__ preb,
                      const float* __restrict__ bond, const int* __restrict__ edge_feat,
                      const int* __restrict__ length, float* __restrict__ aggr) {
  int jt = blockIdx.x, b = blockIdx.y;
  int t = threadIdx.x;
  int j0 = jt * 16;
  __shared__ half_t Wsh[HH * HH];
  __shared__ half_t As[16 * HH];
  __shared__ float bondsh[10 * HH];
  __shared__ float xrow[HH];
  __shared__ float pbsh[HH];
  __shared__ int efsh[16];
  for (int i = t; i < HH * HH; i += 256) Wsh[i] = (half_t)preW[i];
  for (int i = t; i < 10 * HH; i += 256) bondsh[i] = bond[i];
  if (t < HH) pbsh[t] = preb[t];
  __syncthreads();
  int wv = t >> 5, lane = t & 31;
  int n0 = wv * 16;
  int col = lane & 15, rb = (lane >> 4) * 8;
  int h_lane = n0 + col;
  float pb = pbsh[h_lane];
  int len = length[b];
  v8f acc = {};
  for (int i = 0; i < len; ++i) {
    __syncthreads();
    if (t < HH) xrow[t] = xf[((size_t)(b * NN) + i) * HH + t];
    if (t < 16) efsh[t] = edge_feat[((size_t)(b * NN) + i) * NN + j0 + t];
    __syncthreads();
    for (int idx = t; idx < 16 * HH; idx += 256) {
      int row = idx >> 7, c = idx & 127;
      As[idx] = (half_t)(xrow[c] + bondsh[efsh[row] * HH + c]);
    }
    __syncthreads();
    v8f tmp = {};
#pragma unroll
    for (int kc = 0; kc < 4; ++kc) {
      v16h a  = frag_ld(&As[kc * 32], HH, lane);
      v16h bf = frag_ld(&Wsh[n0 * HH + kc * 32], HH, lane);
      tmp = __builtin_amdgcn_wmma_f32_16x16x32_f16(false, a, false, bf, (short)0, tmp,
                                                   false, false);
    }
    const half_t* brow = bases + ((size_t)((b * NN) + i) * NN + j0) * HH;
#pragma unroll
    for (int r = 0; r < 8; ++r) {
      int jg = j0 + rb + r;
      if (jg < len)
        acc[r] += geluf(tmp[r] + pb) * (float)brow[(size_t)(rb + r) * HH + h_lane];
    }
  }
#pragma unroll
  for (int r = 0; r < 8; ++r)
    aggr[((size_t)(b * NN) + j0 + rb + r) * HH + h_lane] = acc[r];
}

// ---------------- misc elementwise / reductions ----------------

__global__ void k_gather_xf(const int* __restrict__ nf, const float* __restrict__ emb,
                            float* __restrict__ xf) {
  int row = blockIdx.x, h = threadIdx.x;
  xf[row * HH + h] = emb[nf[row] * HH + h];
}

__global__ void k_add(float* __restrict__ x, const float* __restrict__ a) {
  int idx = blockIdx.x * HH + threadIdx.x;
  x[idx] += a[idx];
}

// masked batch stats over valid nodes          grid: 1, block: 128
__global__ void k_bnstats(const float* __restrict__ y, const int* __restrict__ length,
                          float* __restrict__ mu, float* __restrict__ ivar) {
  int h = threadIdx.x;
  float s = 0.f, sq = 0.f;
  int cnt = 0;
  for (int b = 0; b < BB; ++b) {
    int len = length[b];
    cnt += len;
    for (int n = 0; n < len; ++n) {
      float v = y[((size_t)(b * NN) + n) * HH + h];
      s += v;
      sq += v * v;
    }
  }
  float m = s / (float)cnt;
  mu[h] = m;
  ivar[h] = rsqrtf(sq / (float)cnt - m * m + EPSF);
}

__global__ void k_bnres(const float* __restrict__ y, const float* __restrict__ mu,
                        const float* __restrict__ ivar, const float* __restrict__ g,
                        const float* __restrict__ bb, float* __restrict__ xf) {
  int row = blockIdx.x, h = threadIdx.x;
  float v = (y[row * HH + h] - mu[h]) * ivar[h] * g[h] + bb[h];
  xf[row * HH + h] += fmaxf(v, 0.f);
}

// masked mean pool + final linear (NC=1)       grid: B, block: 128
__global__ void k_pool(const float* __restrict__ xf, const int* __restrict__ length,
                       const float* __restrict__ linW, const float* __restrict__ linb,
                       float* __restrict__ out) {
  int b = blockIdx.x, h = threadIdx.x;
  __shared__ float red[HH];
  int len = length[b];
  float s = 0.f;
  for (int n = 0; n < len; ++n) s += xf[((size_t)(b * NN) + n) * HH + h];
  red[h] = (s / (float)len) * linW[h];
  __syncthreads();
  for (int st = 64; st > 0; st >>= 1) { if (h < st) red[h] += red[h + st]; __syncthreads(); }
  if (h == 0) out[b] = red[0] + linb[0];
}

// ---------------- launcher ----------------

extern "C" void kernel_launch(void* const* d_in, const int* in_sizes, int n_in,
                              void* d_out, int out_size, void* d_ws, size_t ws_size,
                              hipStream_t stream) {
  const float* e        = (const float*)d_in[0];
  const float* u        = (const float*)d_in[1];
  const float* atom_emb = (const float*)d_in[2];
  const float* bond_emb = (const float*)d_in[3];
  const float* eigw_W   = (const float*)d_in[4];
  const float* eigw_b   = (const float*)d_in[5];
  const float* mha_g    = (const float*)d_in[6];
  const float* mha_b    = (const float*)d_in[7];
  const float* qkv_W    = (const float*)d_in[8];
  const float* qkv_b    = (const float*)d_in[9];
  const float* out_W    = (const float*)d_in[10];
  const float* out_b    = (const float*)d_in[11];
  const float* ffn_g    = (const float*)d_in[12];
  const float* ffn_b    = (const float*)d_in[13];
  const float* ffn1_W   = (const float*)d_in[14];
  const float* ffn1_b   = (const float*)d_in[15];
  const float* ffn2_W   = (const float*)d_in[16];
  const float* ffn2_b   = (const float*)d_in[17];
  const float* dec_W    = (const float*)d_in[18];
  const float* dec_b    = (const float*)d_in[19];
  const float* fe1_W    = (const float*)d_in[20];
  const float* fe1_b    = (const float*)d_in[21];
  const float* fe2_W    = (const float*)d_in[22];
  const float* fe2_b    = (const float*)d_in[23];
  const float* pre_W    = (const float*)d_in[24];
  const float* pre_b    = (const float*)d_in[25];
  const float* f1_W     = (const float*)d_in[26];
  const float* f1_b     = (const float*)d_in[27];
  const float* bn1_g    = (const float*)d_in[28];
  const float* bn1_b    = (const float*)d_in[29];
  const float* f2_W     = (const float*)d_in[30];
  const float* f2_b     = (const float*)d_in[31];
  const float* bn2_g    = (const float*)d_in[32];
  const float* bn2_b    = (const float*)d_in[33];
  const float* lin_W    = (const float*)d_in[34];
  const float* lin_b    = (const float*)d_in[35];
  const int* node_feat  = (const int*)d_in[36];
  const int* edge_feat  = (const int*)d_in[37];
  const int* length     = (const int*)d_in[38];

  float* outp = (float*)d_out;
  float* out_h    = outp;          // [B]   (NC=1)
  float* out_newe = outp + 8;      // [B,NH,N]
  float* out_attn = outp + 4104;   // [B,NH,N,N]

  float* W    = (float*)d_ws;
  float* eig  = W + 0;
  float* xm   = W + 131072;
  float* qkv  = W + 262144;        // 1024*384
  float* ctx  = W + 655360;
  float* t2   = W + 786432;
  float* xf   = W + 917504;
  float* aggr = W + 1048576;
  float* y    = W + 1179648;
  float* y2   = W + 1310720;
  float* mu1  = W + 1441792;
  float* iv1  = mu1 + 128;
  float* mu2  = iv1 + 128;
  float* iv2  = mu2 + 128;
  float* filt = W + 1442816;                       // B*N*N*NH floats
  half_t* bases = (half_t*)(W + 1967104);          // B*N*N*H halves (32MB)

  // spectral encoder (node-level)
  k_sine_eig<<<BB * NN, HH, 0, stream>>>(e, eigw_W, eigw_b, eig);
  k_ln<<<BB * NN, HH, 0, stream>>>(eig, mha_g, mha_b, xm);
  k_rowgemm<<<BB * NN, HH, 0, stream>>>(xm, qkv_W, qkv_b, nullptr, qkv, 3 * HH, 0,
                                        nullptr, nullptr, nullptr, nullptr);
  k_attn<<<dim3(NHEAD, BB), HH, 0, stream>>>(qkv, length, out_attn, ctx);
  k_rowgemm<<<BB * NN, HH, 0, stream>>>(ctx, out_W, out_b, eig, eig, HH, 0,
                                        nullptr, nullptr, nullptr, nullptr);
  k_ln<<<BB * NN, HH, 0, stream>>>(eig, ffn_g, ffn_b, xm);
  k_rowgemm<<<BB * NN, HH, 0, stream>>>(xm, ffn1_W, ffn1_b, nullptr, t2, HH, 1,
                                        nullptr, nullptr, nullptr, nullptr);
  k_rowgemm<<<BB * NN, HH, 0, stream>>>(t2, ffn2_W, ffn2_b, eig, eig, HH, 0,
                                        nullptr, nullptr, nullptr, nullptr);
  k_dec<<<BB, NN, 0, stream>>>(eig, dec_W, dec_b, out_newe);

  // spectral bases (WMMA)
  k_filters<<<dim3(8, 8, BB * NHEAD), 32, 0, stream>>>(u, out_newe, filt);
  k_bases<<<BB * NN * NN / 16, 256, 0, stream>>>(filt, fe1_W, fe1_b, fe2_W, fe2_b, bases);

  // message passing
  k_gather_xf<<<BB * NN, HH, 0, stream>>>(node_feat, atom_emb, xf);
  for (int l = 0; l < LL; ++l) {
    k_msg<<<dim3(8, BB), 256, 0, stream>>>(xf, bases, pre_W + l * HH * HH,
                                           pre_b + l * HH, bond_emb, edge_feat,
                                           length, aggr);
    k_add<<<BB * NN, HH, 0, stream>>>(xf, aggr);
    k_rowgemm<<<BB * NN, HH, 0, stream>>>(xf, f1_W + l * HH * HH, f1_b + l * HH,
                                          nullptr, y, HH, 0,
                                          nullptr, nullptr, nullptr, nullptr);
    k_bnstats<<<1, HH, 0, stream>>>(y, length, mu1, iv1);
    k_rowgemm<<<BB * NN, HH, 0, stream>>>(y, f2_W + l * HH * HH, f2_b + l * HH,
                                          nullptr, y2, HH, 0,
                                          mu1, iv1, bn1_g + l * HH, bn1_b + l * HH);
    k_bnstats<<<1, HH, 0, stream>>>(y2, length, mu2, iv2);
    k_bnres<<<BB * NN, HH, 0, stream>>>(y2, mu2, iv2, bn2_g + l * HH, bn2_b + l * HH, xf);
  }

  k_pool<<<BB, HH, 0, stream>>>(xf, length, lin_W, lin_b, out_h);
}
